// MultiheadAttention_43044162240970
// MI455X (gfx1250) — compile-verified, matrix operand order checked by reading, not run on hardware
//
#include <hip/hip_runtime.h>
#include <stdint.h>

// ---------------------------------------------------------------------------
// MultiheadAttention forward for MI455X (gfx1250), wave32, WMMA bf16.
// B=2, N=2048, D=1024, H=16, HD=64.
// d_out = [ out: B*N*D f32 | A: B*H*N*N f32 ]
// ---------------------------------------------------------------------------

constexpr int Bc = 2;
constexpr int Nc = 2048;
constexpr int Dc = 1024;
constexpr int Hc = 16;
constexpr int HDc = 64;
constexpr int Mrows = Bc * Nc;          // 4096 rows for the projection GEMMs

typedef __attribute__((ext_vector_type(16))) __bf16 v16bf;
typedef __attribute__((ext_vector_type(8)))  float  v8f;

// POD 16B/8B vectors (HIP's uint4/float4 have ctors -> not union-friendly)
struct alignas(16) U4 { uint32_t a[4]; };
struct alignas(8)  U2 { uint32_t a[2]; };
struct alignas(16) F4 { float x, y, z, w; };

union BFrag {            // one 16x32 bf16 WMMA operand (8 VGPRs / lane)
    v16bf v;
    U4 q[2];
    unsigned short us[16];
};

static __device__ __forceinline__ unsigned short f2bf(float f) {
    union { float f; uint32_t u; } x; x.f = f;
    uint32_t u = x.u;
    uint32_t r = ((u >> 16) & 1u) + 0x7FFFu;   // round-to-nearest-even
    return (unsigned short)((u + r) >> 16);
}

static __device__ __forceinline__ v8f wmma_bf16(const v16bf& a, const v16bf& b, v8f c) {
    return __builtin_amdgcn_wmma_f32_16x16x32_bf16(false, a, false, b, (short)0, c,
                                                   false, false);
}

// gfx1250 async global->LDS copy (ASYNCcnt-tracked, bypasses VGPRs).
// LDS dest address = low 32 bits of the generic shared pointer.
static __device__ __forceinline__ void async_ld_b128(void* lds, const void* gptr) {
    uint32_t loff = (uint32_t)(uintptr_t)lds;
    asm volatile("global_load_async_to_lds_b128 %0, %1, off"
                 :: "v"(loff), "v"((uint64_t)(uintptr_t)gptr)
                 : "memory");
}
static __device__ __forceinline__ void wait_async0() {
    asm volatile("s_wait_asynccnt 0x0" ::: "memory");
}

// ---------------------------------------------------------------------------
// GEMM: C[M, 1024] = A[M, 1024] @ W[1024, 1024]^T + bias
// ABF16:  0 -> A is f32 (converted to bf16 while filling LDS), 1 -> A is bf16
// OHEADS: 1 -> store bf16 into [B,H,N,64] head layout, 0 -> store f32 [M,1024]
// Block: 256 thr (8 waves as 4x2), tile 128(M) x 64(N), K step 32.
// ---------------------------------------------------------------------------
template <int ABF16, int OHEADS>
__global__ __launch_bounds__(256) void gemm_bias(
    const void* __restrict__ Av, const float* __restrict__ W,
    const float* __restrict__ bias, void* __restrict__ Ov)
{
    constexpr int K = 1024, NO = 1024;
    __shared__ unsigned short At[128 * 32];
    __shared__ unsigned short Wt[64 * 32];

    const int t    = threadIdx.x;
    const int lane = t & 31;
    const int w    = t >> 5;
    const int ln   = lane & 15;
    const int h8   = (lane >> 4) * 8;
    const int m0   = blockIdx.x * 128;
    const int n0   = blockIdx.y * 64;
    const int wm   = (w >> 1) * 32;       // wave M offset in tile
    const int wn   = (w & 1) * 32;        // wave N offset in tile

    v8f acc[2][2] = {};

    for (int k0 = 0; k0 < K; k0 += 32) {
        // ---- stage A tile (128x32) as bf16 ----
        if (ABF16) {
            const unsigned short* A = (const unsigned short*)Av;
            for (int it = 0; it < 4; ++it) {
                int e = (t + it * 256) * 4;               // 4-elem groups
                int row = e >> 5, col = e & 31;
                U2 d = *(const U2*)(A + (size_t)(m0 + row) * K + k0 + col);
                *(U2*)(&At[row * 32 + col]) = d;
            }
        } else {
            const float* A = (const float*)Av;
            for (int it = 0; it < 4; ++it) {
                int e = (t + it * 256) * 4;
                int row = e >> 5, col = e & 31;
                F4 d = *(const F4*)(A + (size_t)(m0 + row) * K + k0 + col);
                union { unsigned short us[4]; U2 u; } p;
                p.us[0] = f2bf(d.x); p.us[1] = f2bf(d.y);
                p.us[2] = f2bf(d.z); p.us[3] = f2bf(d.w);
                *(U2*)(&At[row * 32 + col]) = p.u;
            }
        }
        // ---- stage W tile (64x32) as bf16; B[k][n] = W[n][k] so lane reads row n
        for (int it = 0; it < 2; ++it) {
            int e = (t + it * 256) * 4;
            int row = e >> 5, col = e & 31;
            F4 d = *(const F4*)(W + (size_t)(n0 + row) * K + k0 + col);
            union { unsigned short us[4]; U2 u; } p;
            p.us[0] = f2bf(d.x); p.us[1] = f2bf(d.y);
            p.us[2] = f2bf(d.z); p.us[3] = f2bf(d.w);
            *(U2*)(&Wt[row * 32 + col]) = p.u;
        }
        __syncthreads();

        BFrag a[2], bfr[2];
        for (int mi = 0; mi < 2; ++mi) {
            int r = wm + mi * 16 + ln;
            a[mi].q[0] = *(const U4*)(&At[r * 32 + h8]);
            a[mi].q[1] = *(const U4*)(&At[r * 32 + h8 + 16]);
        }
        for (int ni = 0; ni < 2; ++ni) {
            int r = wn + ni * 16 + ln;
            bfr[ni].q[0] = *(const U4*)(&Wt[r * 32 + h8]);
            bfr[ni].q[1] = *(const U4*)(&Wt[r * 32 + h8 + 16]);
        }
        for (int mi = 0; mi < 2; ++mi)
            for (int ni = 0; ni < 2; ++ni)
                acc[mi][ni] = wmma_bf16(a[mi].v, bfr[ni].v, acc[mi][ni]);
        __syncthreads();
    }

    // ---- epilogue: +bias, store ----
    for (int mi = 0; mi < 2; ++mi)
        for (int ni = 0; ni < 2; ++ni) {
            int n = n0 + wn + ni * 16 + ln;          // output column (same for all 8)
            float bv = bias[n];
            int mbase = m0 + wm + mi * 16 + h8;      // D-frag: M = (lane/16)*8 + r
            if (OHEADS) {
                unsigned short* O = (unsigned short*)Ov;
                int hh = n >> 6, hd = n & 63;
                for (int r = 0; r < 8; ++r) {
                    int m = mbase + r;
                    int bb = m >> 11, i = m & (Nc - 1);
                    O[((size_t)((bb * Hc + hh) * Nc + i)) * HDc + hd] =
                        f2bf(acc[mi][ni][r] + bv);
                }
            } else {
                float* O = (float*)Ov;
                for (int r = 0; r < 8; ++r)
                    O[(size_t)(mbase + r) * NO + n] = acc[mi][ni][r] + bv;
            }
        }
}

// ---------------------------------------------------------------------------
// Fused attention per (b, h, 64-row Q block). 128 threads = 4 waves,
// each wave owns 16 Q rows. Computes S^T = K @ Q^T so each lane's D-frag
// holds a single Q row -> per-lane softmax stats (one shfl_xor(16) combine).
// Pass 1: online row max / sumexp, K chunks double-buffered via
// global_load_async_to_lds_b128. Pass 2: recompute S (K/V are L2-hot),
// write normalized P to A (f32), repack P D-frags directly as bf16 A-frags
// (layouts match bit-for-bit) and accumulate O = P @ V via WMMA; V is
// software-pipelined through registers and scatter-transposed into LDS.
// ---------------------------------------------------------------------------
__global__ __launch_bounds__(128) void attn_fused(
    const unsigned short* __restrict__ Qh, const unsigned short* __restrict__ Kh,
    const unsigned short* __restrict__ Vh, float* __restrict__ Aout,
    unsigned short* __restrict__ Oout)
{
    __shared__ unsigned short Kt[2][64 * 64];   // key chunk,  [key][hd], dbl-buf
    __shared__ unsigned short Vt[64 * 64];      // val chunk,  [hd][key] (transposed)
    __shared__ float msm[64], lsm[64];

    const int t    = threadIdx.x;
    const int lane = t & 31;
    const int w    = t >> 5;
    const int ln   = lane & 15;
    const int h8   = (lane >> 4) * 8;
    const int qb   = blockIdx.x;
    const int hd_h = blockIdx.y;             // head
    const int b    = blockIdx.z;
    const int bh   = b * Hc + hd_h;
    const int q0   = qb * 64;
    const float scale = 0.125f;              // 1/sqrt(64)
    constexpr int NCH = Nc / 64;             // 32 key chunks

    const unsigned short* Kbase = Kh + (size_t)bh * Nc * HDc;
    const unsigned short* Vbase = Vh + (size_t)bh * Nc * HDc;

    // Q fragments (B-operand of S^T): lane ln holds Q row q0+w*16+ln, hd packed.
    BFrag qf[2];
    {
        const unsigned short* qp = Qh + ((size_t)bh * Nc + q0 + w * 16 + ln) * HDc;
        for (int c = 0; c < 2; ++c) {
            qf[c].q[0] = *(const U4*)(qp + c * 32 + h8);
            qf[c].q[1] = *(const U4*)(qp + c * 32 + 16 + h8);
        }
    }

    // ---------------- pass 1: softmax statistics ----------------
    float mrun = -1e30f, lrun = 0.f;

    // prefetch chunk 0 into buffer 0 (async, LDS-direct)
    for (int it = 0; it < 4; ++it) {
        int off = (t + it * 128) * 8;
        async_ld_b128(&Kt[0][off], Kbase + off);
    }
    wait_async0();
    __syncthreads();

    for (int c = 0; c < NCH; ++c) {
        const int cur = c & 1;
        if (c + 1 < NCH) {                      // prefetch next chunk
            const unsigned short* kp = Kbase + (size_t)(c + 1) * 64 * HDc;
            for (int it = 0; it < 4; ++it) {
                int off = (t + it * 128) * 8;
                async_ld_b128(&Kt[cur ^ 1][off], kp + off);
            }
        }

        float cmax = -1e30f;
        v8f s[4];
        for (int f = 0; f < 4; ++f) {
            v8f sa = {};
            int r = f * 16 + ln;
            for (int cc = 0; cc < 2; ++cc) {
                BFrag kf;
                kf.q[0] = *(const U4*)(&Kt[cur][r * 64 + cc * 32 + h8]);
                kf.q[1] = *(const U4*)(&Kt[cur][r * 64 + cc * 32 + 16 + h8]);
                sa = wmma_bf16(kf.v, qf[cc].v, sa);
            }
            for (int rr = 0; rr < 8; ++rr) {
                float v = sa[rr] * scale;
                s[f][rr] = v;
                cmax = fmaxf(cmax, v);
            }
        }
        float mnew = fmaxf(mrun, cmax);
        float add = 0.f;
        for (int f = 0; f < 4; ++f)
            for (int rr = 0; rr < 8; ++rr) add += __expf(s[f][rr] - mnew);
        lrun = lrun * __expf(mrun - mnew) + add;
        mrun = mnew;

        if (c + 1 < NCH) wait_async0();
        __syncthreads();
    }

    // combine the two key-halves (lane and lane^16 share the same Q row)
    {
        float mo = __shfl_xor(mrun, 16, 32);
        float lo = __shfl_xor(lrun, 16, 32);
        float mrow = fmaxf(mrun, mo);
        float lrow = lrun * __expf(mrun - mrow) + lo * __expf(mo - mrow);
        if (lane < 16) { msm[w * 16 + lane] = mrow; lsm[w * 16 + lane] = lrow; }
    }
    __syncthreads();
    const float mrow = msm[w * 16 + ln];
    const float invl = 1.f / lsm[w * 16 + ln];

    // ---------------- pass 2: write A, accumulate O = P @ V ----------------
    v8f oacc[4] = {};
    float* arow = Aout + ((size_t)bh * Nc + q0 + w * 16 + ln) * Nc;

    // prefetch K chunk 0 (async->LDS) and V chunk 0 (into registers)
    U4 vcur[4], vnxt[4];
    for (int it = 0; it < 4; ++it) {
        int off = (t + it * 128) * 8;
        async_ld_b128(&Kt[0][off], Kbase + off);
        vcur[it] = *(const U4*)(Vbase + off);
    }
    wait_async0();
    __syncthreads();

    for (int c = 0; c < NCH; ++c) {
        const int cur = c & 1;

        // scatter-transpose V chunk c from registers into Vt[hd][key]
        for (int it = 0; it < 4; ++it) {
            int e8 = t + it * 128;
            union { U4 q; unsigned short us[8]; } vv; vv.q = vcur[it];
            int vr = e8 >> 3, vd0 = (e8 & 7) * 8;
            for (int i = 0; i < 8; ++i) Vt[(vd0 + i) * 64 + vr] = vv.us[i];
        }
        // issue next chunk's K (async->LDS) and V (global->VGPR) fetches
        if (c + 1 < NCH) {
            const unsigned short* kp = Kbase + (size_t)(c + 1) * 64 * HDc;
            const unsigned short* vp = Vbase + (size_t)(c + 1) * 64 * HDc;
            for (int it = 0; it < 4; ++it) {
                int off = (t + it * 128) * 8;
                async_ld_b128(&Kt[cur ^ 1][off], kp + off);
                vnxt[it] = *(const U4*)(vp + off);
            }
        }
        __syncthreads();   // Vt (and this wave's Kt[cur] reads) ready

        v8f p[4];
        for (int f = 0; f < 4; ++f) {
            v8f sa = {};
            int r = f * 16 + ln;
            for (int cc = 0; cc < 2; ++cc) {
                BFrag kf;
                kf.q[0] = *(const U4*)(&Kt[cur][r * 64 + cc * 32 + h8]);
                kf.q[1] = *(const U4*)(&Kt[cur][r * 64 + cc * 32 + 16 + h8]);
                sa = wmma_bf16(kf.v, qf[cc].v, sa);
            }
            for (int rr = 0; rr < 8; ++rr)
                p[f][rr] = __expf(sa[rr] * scale - mrow) * invl;
            // store normalized probabilities (key = c*64 + f*16 + h8 + rr)
            float* ap = arow + c * 64 + f * 16 + h8;
            union { F4 f4; float fl[4]; } o0, o1;
            for (int rr = 0; rr < 4; ++rr) { o0.fl[rr] = p[f][rr]; o1.fl[rr] = p[f][4 + rr]; }
            *(F4*)(ap)     = o0.f4;
            *(F4*)(ap + 4) = o1.f4;
        }

        // P D-frags -> bf16 A-frags (element order matches exactly, no shuffles)
        BFrag pa01, pa23;
        for (int rr = 0; rr < 8; ++rr) {
            pa01.us[rr]     = f2bf(p[0][rr]);
            pa01.us[8 + rr] = f2bf(p[1][rr]);
            pa23.us[rr]     = f2bf(p[2][rr]);
            pa23.us[8 + rr] = f2bf(p[3][rr]);
        }
        for (int fc = 0; fc < 4; ++fc) {
            int r = fc * 16 + ln;
            BFrag v0, v1;
            v0.q[0] = *(const U4*)(&Vt[r * 64 + h8]);
            v0.q[1] = *(const U4*)(&Vt[r * 64 + 16 + h8]);
            v1.q[0] = *(const U4*)(&Vt[r * 64 + 32 + h8]);
            v1.q[1] = *(const U4*)(&Vt[r * 64 + 32 + 16 + h8]);
            oacc[fc] = wmma_bf16(pa01.v, v0.v, oacc[fc]);
            oacc[fc] = wmma_bf16(pa23.v, v1.v, oacc[fc]);
        }

        if (c + 1 < NCH) wait_async0();
        __syncthreads();   // everyone done with Kt[cur] / Vt before overwrite
        for (int it = 0; it < 4; ++it) vcur[it] = vnxt[it];
    }

    // store O (bf16, [B,N,D] row-major) for the output projection
    for (int fc = 0; fc < 4; ++fc) {
        int hd = fc * 16 + ln;
        for (int rr = 0; rr < 8; ++rr) {
            int qr = q0 + w * 16 + h8 + rr;
            Oout[((size_t)b * Nc + qr) * Dc + hd_h * HDc + hd] = f2bf(oacc[fc][rr]);
        }
    }
}

// ---------------------------------------------------------------------------
extern "C" void kernel_launch(void* const* d_in, const int* in_sizes, int n_in,
                              void* d_out, int out_size, void* d_ws, size_t ws_size,
                              hipStream_t stream) {
    const float* query = (const float*)d_in[0];
    const float* key_  = (const float*)d_in[1];
    const float* value = (const float*)d_in[2];
    const float* Wq = (const float*)d_in[3];
    const float* bq = (const float*)d_in[4];
    const float* Wk = (const float*)d_in[5];
    const float* bk = (const float*)d_in[6];
    const float* Wv = (const float*)d_in[7];
    const float* bv = (const float*)d_in[8];
    const float* Wo = (const float*)d_in[9];
    const float* bo = (const float*)d_in[10];

    constexpr size_t HEAD_ELEMS = (size_t)Bc * Hc * Nc * HDc;   // 4,194,304
    unsigned short* Qh = (unsigned short*)d_ws;
    unsigned short* Kh = Qh + HEAD_ELEMS;
    unsigned short* Vh = Kh + HEAD_ELEMS;
    unsigned short* VO = Vh + HEAD_ELEMS;   // attention output, bf16 [B*N, D]

    float* out  = (float*)d_out;                       // [B*N, D] f32
    float* Aout = out + (size_t)Bc * Nc * Dc;          // [B,H,N,N] f32

    dim3 gg(Mrows / 128, Dc / 64);                     // 32 x 16 blocks

    gemm_bias<0, 1><<<gg, 256, 0, stream>>>(query, Wq, bq, Qh);
    gemm_bias<0, 1><<<gg, 256, 0, stream>>>(key_,  Wk, bk, Kh);
    gemm_bias<0, 1><<<gg, 256, 0, stream>>>(value, Wv, bv, Vh);

    attn_fused<<<dim3(Nc / 64, Hc, Bc), 128, 0, stream>>>(Qh, Kh, Vh, Aout, VO);

    gemm_bias<1, 0><<<gg, 256, 0, stream>>>(VO, Wo, bo, out);
}
